// EdgePerturbationPredictor_90658169684335
// MI455X (gfx1250) — matrix operand order repaired and checked by reading, not compiled, not run on hardware
//
#include <hip/hip_runtime.h>

typedef __attribute__((ext_vector_type(16))) _Float16 v16h;
typedef __attribute__((ext_vector_type(8)))  float    v8f;

#define FEAT 128
#define HID  64
#define HID2 32
#define WAVES_PER_BLOCK 8
#define THREADS (WAVES_PER_BLOCK * 32)
#define NBLOCKS 1024
#define HROW 80   // f16 row stride (160B): 16B aligned, bank-staggered

union FragU { uint4 u[2]; v16h h; };

static __device__ __forceinline__ unsigned int pk_f16(float a, float b) {
  union { _Float16 h[2]; unsigned int u; } x;
  x.h[0] = (_Float16)a; x.h[1] = (_Float16)b;
  return x.u;
}

__global__ __launch_bounds__(THREADS)
void edge_mlp_wmma(const float* __restrict__ nf,
                   const long long* __restrict__ ei,
                   const float* __restrict__ W1, const float* __restrict__ b1,
                   const float* __restrict__ W2, const float* __restrict__ b2,
                   const float* __restrict__ W3, const float* __restrict__ b3,
                   float* __restrict__ out, int nEdges)
{
  // B fragments of W1 (256x64) and W2 (64x32), f16, in WMMA B-layout:
  //   idx = ((chunk*NT + ntile)*2 + h)*32 + lane ; each entry = uint4 (8 f16 = 4 VGPRs)
  __shared__ uint4 w1frag[8 * 4 * 2 * 32];   // 32 KB
  __shared__ uint4 w2frag[2 * 2 * 2 * 32];   // 4 KB
  __shared__ __align__(16) _Float16 hbuf[WAVES_PER_BLOCK * 16 * HROW]; // 20 KB

  const int tid = threadIdx.x;

  // ---- one-time per block: stage W1/W2 as f16 B-fragments in LDS ----
  for (int idx = tid; idx < 8 * 4 * 2 * 32; idx += THREADS) {
    const int lane = idx & 31;
    const int h    = (idx >> 5) & 1;
    const int t    = (idx >> 6) & 3;
    const int c    = idx >> 8;                       // K-chunk 0..7
    const int n     = t * 16 + (lane & 15);          // column
    const int kbase = c * 32 + (lane >> 4) * 8 + h * 16;
    uint4 u;
    unsigned int* up = (unsigned int*)&u;
#pragma unroll
    for (int i = 0; i < 4; ++i)
      up[i] = pk_f16(W1[(kbase + 2 * i) * HID + n],
                     W1[(kbase + 2 * i + 1) * HID + n]);
    w1frag[idx] = u;
  }
  for (int idx = tid; idx < 2 * 2 * 2 * 32; idx += THREADS) {
    const int lane = idx & 31;
    const int h    = (idx >> 5) & 1;
    const int t    = (idx >> 6) & 1;
    const int c    = idx >> 7;                       // K-chunk 0..1
    const int n     = t * 16 + (lane & 15);
    const int kbase = c * 32 + (lane >> 4) * 8 + h * 16;
    uint4 u;
    unsigned int* up = (unsigned int*)&u;
#pragma unroll
    for (int i = 0; i < 4; ++i)
      up[i] = pk_f16(W2[(kbase + 2 * i) * HID2 + n],
                     W2[(kbase + 2 * i + 1) * HID2 + n]);
    w2frag[idx] = u;
  }
  __syncthreads();

  const int lane   = tid & 31;
  const int waveId = tid >> 5;
  const int m = lane & 15;          // A: edge-row ; C: column-within-N-tile
  const int g = lane >> 4;          // lane group
  const int globalWave = blockIdx.x * WAVES_PER_BLOCK + waveId;
  const int nWaves = gridDim.x * WAVES_PER_BLOCK;

  _Float16* myh  = &hbuf[waveId * 16 * HROW];
  float*    myh2 = (float*)myh;     // reused as f32 [16][40]

  // ---- hoist loop-invariants: W2 fragments -> registers, biases ----
  FragU w2reg[2][2];                // [chunk][ntile], 32 VGPRs total
#pragma unroll
  for (int c = 0; c < 2; ++c)
#pragma unroll
    for (int t = 0; t < 2; ++t) {
      const int base = ((c * 2 + t) * 2) * 32;
      w2reg[c][t].u[0] = w2frag[base + lane];
      w2reg[c][t].u[1] = w2frag[base + 32 + lane];
    }
  float bias1[4], bias2[2];
#pragma unroll
  for (int t = 0; t < 4; ++t) bias1[t] = b1[t * 16 + m];
#pragma unroll
  for (int t = 0; t < 2; ++t) bias2[t] = b2[t * 16 + m];
  const float bias3 = b3[0];

  const int nTiles = (nEdges + 15) >> 4;
  for (int tile = globalWave; tile < nTiles; tile += nWaves) {
    int e = tile * 16 + m;
    if (e >= nEdges) e = nEdges - 1;              // safe clamp for tail tile

    // prefetch next tile's edge indices (global_prefetch_b8)
    const int nextTile = tile + nWaves;
    if (nextTile < nTiles) {
      int en = nextTile * 16 + m;
      if (en >= nEdges) en = nEdges - 1;
      __builtin_prefetch(ei + en, 0, 3);
      __builtin_prefetch(ei + (size_t)nEdges + en, 0, 3);
    }

    const long long sidx = ei[e];
    const long long didx = ei[(size_t)nEdges + e];
    const float* srow = nf + (size_t)sidx * FEAT; // L2-resident gathers
    const float* drow = nf + (size_t)didx * FEAT;

    // ---- gather + cvt A fragments: 16 edges x 256 feats, A-layout ----
    v16h a[8];
#pragma unroll
    for (int c = 0; c < 8; ++c) {
      const int k0 = c * 32 + g * 8;              // elems 0..7
      const int k1 = k0 + 16;                     // elems 8..15
      const float* p0 = (k0 < FEAT) ? (srow + k0) : (drow + (k0 - FEAT));
      const float* p1 = (k1 < FEAT) ? (srow + k1) : (drow + (k1 - FEAT));
      float4 f0 = *(const float4*)(p0);
      float4 f1 = *(const float4*)(p0 + 4);
      float4 f2 = *(const float4*)(p1);
      float4 f3 = *(const float4*)(p1 + 4);
      v16h av;
      av[0]=(_Float16)f0.x;  av[1]=(_Float16)f0.y;  av[2]=(_Float16)f0.z;  av[3]=(_Float16)f0.w;
      av[4]=(_Float16)f1.x;  av[5]=(_Float16)f1.y;  av[6]=(_Float16)f1.z;  av[7]=(_Float16)f1.w;
      av[8]=(_Float16)f2.x;  av[9]=(_Float16)f2.y;  av[10]=(_Float16)f2.z; av[11]=(_Float16)f2.w;
      av[12]=(_Float16)f3.x; av[13]=(_Float16)f3.y; av[14]=(_Float16)f3.z; av[15]=(_Float16)f3.w;
      a[c] = av;
    }

    // ---- layer 1: [16,256] @ [256,64] -> relu -> f16 scratch ----
#pragma unroll
    for (int t = 0; t < 4; ++t) {
      v8f acc = {0.f,0.f,0.f,0.f,0.f,0.f,0.f,0.f};
#pragma unroll
      for (int c = 0; c < 8; ++c) {
        const int base = ((c * 4 + t) * 2) * 32;
        FragU bf;
        bf.u[0] = w1frag[base + lane];
        bf.u[1] = w1frag[base + 32 + lane];
        acc = __builtin_amdgcn_wmma_f32_16x16x32_f16(false, a[c], false, bf.h,
                                                     (short)0, acc, false, false);
      }
#pragma unroll
      for (int r = 0; r < 8; ++r) {
        float v = acc[r] + bias1[t];
        v = v > 0.f ? v : 0.f;
        myh[(r + 8 * g) * HROW + t * 16 + m] = (_Float16)v;  // row=C M-index
      }
    }

    // ---- re-form A fragments for layer 2 (cross-lane via LDS) ----
    v16h a2[2];
#pragma unroll
    for (int c = 0; c < 2; ++c) {
      const _Float16* rp = myh + m * HROW;
      const int k0 = c * 32 + g * 8;
      FragU fa;
      fa.u[0] = *(const uint4*)(rp + k0);
      fa.u[1] = *(const uint4*)(rp + k0 + 16);
      a2[c] = fa.h;
    }

    // ---- layer 2: [16,64] @ [64,32] -> relu -> f32 scratch ----
#pragma unroll
    for (int t = 0; t < 2; ++t) {
      v8f acc = {0.f,0.f,0.f,0.f,0.f,0.f,0.f,0.f};
#pragma unroll
      for (int c = 0; c < 2; ++c) {
        acc = __builtin_amdgcn_wmma_f32_16x16x32_f16(false, a2[c], false,
                                                     w2reg[c][t].h,
                                                     (short)0, acc, false, false);
      }
#pragma unroll
      for (int r = 0; r < 8; ++r) {
        float v = acc[r] + bias2[t];
        v = v > 0.f ? v : 0.f;
        myh2[(r + 8 * g) * 40 + t * 16 + m] = v;
      }
    }

    // ---- layer 3: 32-wide dot + sigmoid (one edge per lane 0..15) ----
    if (lane < 16) {
      const float* hr = myh2 + m * 40;
      float s = bias3;
#pragma unroll
      for (int j = 0; j < HID2; ++j) s += hr[j] * W3[j];
      const float y = 1.f / (1.f + __expf(-s));
      const int eo = tile * 16 + m;
      if (eo < nEdges) out[eo] = y;
    }
  }
}

extern "C" void kernel_launch(void* const* d_in, const int* in_sizes, int n_in,
                              void* d_out, int out_size, void* d_ws, size_t ws_size,
                              hipStream_t stream) {
  const float*     nf = (const float*)d_in[0];
  const long long* ei = (const long long*)d_in[1];
  const float*     W1 = (const float*)d_in[2];
  const float*     b1 = (const float*)d_in[3];
  const float*     W2 = (const float*)d_in[4];
  const float*     b2 = (const float*)d_in[5];
  const float*     W3 = (const float*)d_in[6];
  const float*     b3 = (const float*)d_in[7];
  float* out = (float*)d_out;

  const int nEdges = out_size;                  // E
  const int nTiles = (nEdges + 15) / 16;
  int maxBlocks = (nTiles + WAVES_PER_BLOCK - 1) / WAVES_PER_BLOCK;
  int blocks = NBLOCKS < maxBlocks ? NBLOCKS : maxBlocks;
  if (blocks < 1) blocks = 1;

  edge_mlp_wmma<<<blocks, THREADS, 0, stream>>>(nf, ei, W1, b1, W2, b2, W3, b3,
                                                out, nEdges);
}
